// FFTResonanceBlock_28432683499852
// MI455X (gfx1250) — matrix operand order repaired and checked by reading, not compiled, not run on hardware
//
#include <hip/hip_runtime.h>
#include <hip/hip_bf16.h>

typedef __attribute__((ext_vector_type(16))) _Float16 v16h;
typedef __attribute__((ext_vector_type(8)))  float    v8f;
typedef unsigned int v4u  __attribute__((ext_vector_type(4)));
typedef int          v8i_ __attribute__((ext_vector_type(8)));
typedef int          v4i_ __attribute__((ext_vector_type(4)));

#define WINDOW    2048
#define HOP       1024
#define N_COEFFS  1025
#define EXPR_     4
#define NB        256              // N_RES * EXPR
#define N_FRAMES  128
#define N_SAMPLES 131072
#define M_TOT     (NB * N_FRAMES)  // 32768
#define K_RE      1025
#define K_TOT     2050
#define KP        2080             // K padded to multiple of 32
#define NN        WINDOW           // 2048
#define PI_F      3.14159265358979f

// ---------------------------------------------------------------- zero output
__global__ void zero_f32(float* __restrict__ p, int n) {
  for (int i = blockIdx.x * blockDim.x + threadIdx.x; i < n;
       i += gridDim.x * blockDim.x)
    p[i] = 0.0f;
}

// ------------------------------------------------- DFT basis, transposed [n][k]
__global__ void gen_basis(_Float16* __restrict__ Bt) {
  int idx = blockIdx.x * blockDim.x + threadIdx.x;
  if (idx >= NN * KP) return;
  int n = idx / KP;
  int k = idx % KP;
  const float w = 2.0f * PI_F / (float)WINDOW;
  float v = 0.0f;
  if (k < K_RE) {
    v = __cosf(w * (float)((k * n) & (WINDOW - 1)));
  } else if (k < K_TOT) {
    v = -__sinf(w * (float)(((k - K_RE) * n) & (WINDOW - 1)));
  }
  Bt[idx] = (_Float16)v;
}

// -------------------------------------------- spectrum rows (A matrix, f16)
__global__ void gen_spec(const float* __restrict__ amp,
                         const float* __restrict__ phase,
                         const float* __restrict__ decay,
                         const float* __restrict__ dith,
                         _Float16* __restrict__ A) {
  int idx = blockIdx.x * blockDim.x + threadIdx.x;
  if (idx >= NB * N_COEFFS) return;
  int b = idx / N_COEFFS;
  int c = idx % N_COEFFS;
  int r = b >> 2, e = b & 3;
  int in_idx = (r * N_COEFFS + c) * EXPR_ + e;   // inputs are (N_RES, N_COEFFS, EXPR)

  float a  = amp[in_idx];
  float ph = phase[in_idx];
  float de = decay[in_idx];
  float pd = dith[in_idx];

  float coeff  = 0.5f + (1.0f / (1.0f + __expf(-de))) * 0.5f * 0.99f;
  float mag    = a * a;
  float sphase = tanhf(ph) * PI_F;
  float dt     = tanhf(pd);
  float scale  = (c == 0 || c == N_COEFFS - 1) ? (1.0f / WINDOW) : (2.0f / WINDOW);

  float acc = 0.0f;
  for (int t = 0; t < N_FRAMES; ++t) {
    mag *= coeff;
    uint32_t h = (uint32_t)((b * N_FRAMES + t) * N_COEFFS + c);
    h ^= h >> 16; h *= 0x7feb352dU; h ^= h >> 15; h *= 0x846ca68bU; h ^= h >> 16;
    float noise = (float)(h >> 8) * (2.0f / 16777216.0f) - 1.0f;
    acc += sphase + dt * noise;

    float s, co;
    __sincosf(acc, &s, &co);
    float m = scale * mag;
    size_t row = (size_t)(b * N_FRAMES + t) * KP;
    A[row + c]        = (_Float16)(m * co);
    A[row + K_RE + c] = (_Float16)(m * s);
    if (c < KP - K_TOT) A[row + K_TOT + c] = (_Float16)0.0f;
  }
}

// ---------------------------------------------------- TDM tile load (2-D D#)
// Loads a [128 rows x 32 halves] tile at gptr (row stride KP halves) into LDS.
__device__ __forceinline__ void tdm_load_tile(uint32_t lds_off,
                                              const _Float16* gptr,
                                              uint32_t tensor_rows) {
  uint64_t ga = (uint64_t)(uintptr_t)gptr;
  v4u g0;
  g0.x = 1u;                                               // count=1, load, user D#
  g0.y = lds_off;                                          // lds_addr[31:0]
  g0.z = (uint32_t)ga;                                     // global_addr[31:0]
  g0.w = (uint32_t)((ga >> 32) & 0x1FFFFFFu) | (2u << 30); // ga[56:32] | type=2
  v8i_ g1;
  g1[0] = (int)(1u << 16);                                 // data_size = 2 bytes
  g1[1] = (int)(((uint32_t)KP & 0xFFFFu) << 16);           // tensor_dim0[15:0]
  g1[2] = (int)(((uint32_t)KP >> 16) |
                ((tensor_rows & 0xFFFFu) << 16));          // dim0[31:16] | dim1[15:0]
  g1[3] = (int)((tensor_rows >> 16) | (32u << 16));        // dim1[31:16] | tile_dim0=32
  g1[4] = 128;                                             // tile_dim1=128, tile_dim2=0
  g1[5] = (int)KP;                                         // tensor_dim0_stride[31:0]
  g1[6] = 0;
  g1[7] = 0;
  v4i_ z4 = {0, 0, 0, 0};
#if defined(__clang_major__) && (__clang_major__ >= 23)
  v8i_ z8 = {0, 0, 0, 0, 0, 0, 0, 0};
  __builtin_amdgcn_tensor_load_to_lds(g0, g1, z4, z4, z8, 0);
#else
  __builtin_amdgcn_tensor_load_to_lds(g0, g1, z4, z4, 0);
#endif
}

// --------------------------------------------------------- WMMA GEMM + OLA
// frames[32768,2048] = A[32768,KP] x BasisT^T[KP,2048], overlap-add epilogue.
// WG = 256 threads = 8 wave32; WG tile 128x128; wave tile 32(M) x 64(N).
// Tiles staged in double-buffered LDS by the Tensor Data Mover (wave 0).
__launch_bounds__(256)
__global__ void gemm_wmma_ola(const _Float16* __restrict__ A,
                              const _Float16* __restrict__ Bt,
                              float* __restrict__ out) {
  __shared__ _Float16 As[2][128 * 32];   // 2 x 8 KB
  __shared__ _Float16 Bs[2][128 * 32];   // 2 x 8 KB

  const int lane  = threadIdx.x & 31;
  const int wave  = threadIdx.x >> 5;    // 0..7
  const int wm    = wave & 3;            // 4 waves along M (32 rows each)
  const int wn    = wave >> 2;           // 2 waves along N (64 cols each)
  const int lrow  = lane & 15;
  const int lhalf = lane >> 4;
  const int mtile = blockIdx.x * 128;
  const int ntile = blockIdx.y * 128;

  v8f acc[2][4];
#pragma unroll
  for (int i = 0; i < 2; ++i)
#pragma unroll
    for (int j = 0; j < 4; ++j)
      acc[i][j] = (v8f){0.f, 0.f, 0.f, 0.f, 0.f, 0.f, 0.f, 0.f};

  // prologue: fill buffer 0
  if (wave == 0) {
    tdm_load_tile((uint32_t)(uintptr_t)&As[0][0], A  + (size_t)mtile * KP, M_TOT);
    tdm_load_tile((uint32_t)(uintptr_t)&Bs[0][0], Bt + (size_t)ntile * KP, NN);
  }
  __builtin_amdgcn_s_wait_tensorcnt(0);
  __syncthreads();

  const int NK = KP / 32;                // 65
  for (int ki = 0; ki < NK; ++ki) {
    const int cur = ki & 1;
    const int nxt = cur ^ 1;

    if ((ki + 1 < NK) && (wave == 0)) {  // prefetch next K-slice via TDM
      const int kk = (ki + 1) * 32;
      tdm_load_tile((uint32_t)(uintptr_t)&As[nxt][0],
                    A + (size_t)mtile * KP + kk, M_TOT);
      tdm_load_tile((uint32_t)(uintptr_t)&Bs[nxt][0],
                    Bt + (size_t)ntile * KP + kk, NN);
    }

    // --- A fragments from LDS: 16x32, lane holds M=lrow
    union { v16h h; uint4 q[2]; } af[2];
#pragma unroll
    for (int i = 0; i < 2; ++i) {
      const uint4* p =
          (const uint4*)&As[cur][(wm * 32 + i * 16 + lrow) * 32 + lhalf * 8];
      af[i].q[0] = p[0];                 // K offsets lhalf*8 + 0..7
      af[i].q[1] = p[2];                 // K offsets lhalf*8 + 16..23
    }
    // --- B fragments from LDS: 32x16, lane holds N=lrow, K offsets lhalf*8 + 0..7 and 16..23
    union { v16h h; uint4 q[2]; } bf[4];
#pragma unroll
    for (int j = 0; j < 4; ++j) {
      const uint4* p =
          (const uint4*)&Bs[cur][(wn * 64 + j * 16 + lrow) * 32 + lhalf * 8];
      bf[j].q[0] = p[0];
      bf[j].q[1] = p[2];
    }

#pragma unroll
    for (int i = 0; i < 2; ++i)
#pragma unroll
      for (int j = 0; j < 4; ++j)
        acc[i][j] = __builtin_amdgcn_wmma_f32_16x16x32_f16(
            false, af[i].h, false, bf[j].h, (short)0, acc[i][j], false, false);

    __builtin_amdgcn_s_wait_tensorcnt(0);  // next buffer landed (wave 0)
    __syncthreads();                       // publish + protect cur for reuse
  }

  // ---- epilogue: overlap-add directly into out (hop = 1024)
#pragma unroll
  for (int i = 0; i < 2; ++i)
#pragma unroll
    for (int j = 0; j < 4; ++j)
#pragma unroll
      for (int r2 = 0; r2 < 8; ++r2) {
        float v = acc[i][j][r2];
        int m = mtile + wm * 32 + i * 16 + 8 * lhalf + r2;
        int n = ntile + wn * 64 + j * 16 + lrow;
        int b = m >> 7;
        int t = m & (N_FRAMES - 1);
        if (n < HOP) {
          atomicAdd(&out[(size_t)b * N_SAMPLES + t * HOP + n], v);
        } else if (t != N_FRAMES - 1) {
          atomicAdd(&out[(size_t)b * N_SAMPLES + (t + 1) * HOP + (n - HOP)], v);
        }
      }
}

// ---------------------------------------------------------------- launcher
extern "C" void kernel_launch(void* const* d_in, const int* in_sizes, int n_in,
                              void* d_out, int out_size, void* d_ws, size_t ws_size,
                              hipStream_t stream) {
  (void)in_sizes; (void)n_in; (void)ws_size;
  const float* amp   = (const float*)d_in[0];
  const float* phase = (const float*)d_in[1];
  const float* decay = (const float*)d_in[2];
  const float* dith  = (const float*)d_in[3];
  float* out = (float*)d_out;

  _Float16* A  = (_Float16*)d_ws;                                   // 32768*2080*2 B
  _Float16* Bt = (_Float16*)((char*)d_ws + (size_t)M_TOT * KP * 2); // 2048*2080*2 B

  zero_f32<<<2048, 256, 0, stream>>>(out, out_size);

  {
    int n = NN * KP;
    gen_basis<<<(n + 255) / 256, 256, 0, stream>>>(Bt);
  }
  {
    int n = NB * N_COEFFS;
    gen_spec<<<(n + 255) / 256, 256, 0, stream>>>(amp, phase, decay, dith, A);
  }

  dim3 grid(M_TOT / 128, NN / 128);   // (256, 16)
  gemm_wmma_ola<<<grid, 256, 0, stream>>>(A, Bt, out);
}
